// InteractingLayer_51144470561148
// MI455X (gfx1250) — matrix operand order repaired and checked by reading, not compiled, not run on hardware
//
#include <hip/hip_runtime.h>
#include <hip/hip_bf16.h>

typedef __attribute__((ext_vector_type(16))) _Float16 v16h;
typedef __attribute__((ext_vector_type(8)))  _Float16 v8h;
typedef __attribute__((ext_vector_type(8)))  float    v8f;

#define H_   8
#define N_   64
#define D_   128
#define ATT_ 32
static constexpr float SCALE = 0.17677669529663687f; // 32^-0.5

// LDS layout (element strides chosen for 16B alignment + bank-conflict avoidance)
#define XSTR  136   // halves per row of xh   (row*272B, +16B pad)
#define QSTR  40    // halves per row of Qh/Kh (row*80B)
#define VTSTR 72    // halves per row of Vt    (row*144B)
#define SSTR  68    // floats per row of S     (row*272B)

#define XH_BYTES   (64 * XSTR * 2)                               // 17408
#define QH_BYTES   (64 * QSTR * 2)                               // 5120
#define KH_BYTES   (64 * QSTR * 2)                               // 5120
#define VT_BYTES   (32 * VTSTR * 2)                              // 4608
#define S_BYTES    (16 * SSTR * 4)                               // 4352
#define HEAD_BYTES (QH_BYTES + KH_BYTES + VT_BYTES + S_BYTES)    // 19200
#define SMEM_BYTES (XH_BYTES + 8 * HEAD_BYTES)                   // 171008

union FragH { v16h v; v8h h[2]; _Float16 e[16]; };

// A fragment (16x32 f16) from row-major [M][strideH] storage.
// ISA layout: lane row M=lane&15; element i -> K = (i&7) + 8*(lane>>4) + 16*(i>>3)
__device__ __forceinline__ v16h load_a_f16(const _Float16* base, int strideH,
                                           int M0, int K0, int lane) {
  const _Float16* p = base + (M0 + (lane & 15)) * strideH + K0 + ((lane >> 4) << 3);
  FragH f;
  f.h[0] = *(const v8h*)(p);
  f.h[1] = *(const v8h*)(p + 16);
  return f.v;
}

// B fragment (32x16 f16) from K-contiguous (i.e. B^T row-major) [N][strideH] storage.
// ISA layout: lane col N=lane&15; element i -> K = (i&7) + 8*(lane>>4) + 16*(i>>3)
__device__ __forceinline__ v16h load_b_f16(const _Float16* base, int strideH,
                                           int N0, int K0, int lane) {
  const _Float16* p = base + (N0 + (lane & 15)) * strideH + K0 + ((lane >> 4) << 3);
  FragH f;
  f.h[0] = *(const v8h*)(p);
  f.h[1] = *(const v8h*)(p + 16);
  return f.v;
}

// A fragment from f32 row-major source (convert on load); rows 0..15 of `base`
__device__ __forceinline__ v16h load_a_f32cvt(const float* base, int strideF,
                                              int K0, int lane) {
  const float* p = base + (lane & 15) * strideF + K0 + ((lane >> 4) << 3);
  FragH f;
#pragma unroll
  for (int i = 0; i < 8; ++i) f.e[i] = (_Float16)p[i];
#pragma unroll
  for (int i = 0; i < 8; ++i) f.e[8 + i] = (_Float16)p[16 + i];
  return f.v;
}

__device__ __forceinline__ v8f wmma16(v16h a, v16h b, v8f c) {
  return __builtin_amdgcn_wmma_f32_16x16x32_f16(false, a, false, b,
                                                (short)0, c, false, false);
}

// ---------------------------------------------------------------------------
// Kernel 1: convert weights to f16, transposed to K-contiguous layouts.
//   WQ/WK/WV: (H,D,ATT) f32  -> (H,ATT,D) f16
//   WR:       (D,256)   f32  -> (256,D)   f16
// ---------------------------------------------------------------------------
__global__ __launch_bounds__(256) void prep_weights(
    const float* __restrict__ WQ, const float* __restrict__ WK,
    const float* __restrict__ WV, const float* __restrict__ WR,
    _Float16* __restrict__ oq, _Float16* __restrict__ ok2,
    _Float16* __restrict__ ov, _Float16* __restrict__ orr) {
  const int idx = blockIdx.x * 256 + threadIdx.x;
  if (idx < H_ * ATT_ * D_) {                 // 32768
    const int h = idx / (ATT_ * D_);
    const int n = (idx / D_) % ATT_;
    const int k = idx % D_;
    const int src = (h * D_ + k) * ATT_ + n;
    oq[idx]  = (_Float16)WQ[src];
    ok2[idx] = (_Float16)WK[src];
    ov[idx]  = (_Float16)WV[src];
    const int c = idx / D_;                   // 0..255
    orr[idx] = (_Float16)WR[k * (H_ * ATT_) + c];
  }
}

// ---------------------------------------------------------------------------
// Kernel 2: fused MHA, one batch element per block, one head per wave.
// ---------------------------------------------------------------------------
__global__ __launch_bounds__(256) void attn_fused(
    const float* __restrict__ x,
    const _Float16* __restrict__ wqt,   // [H][ATT][D] f16
    const _Float16* __restrict__ wkt,
    const _Float16* __restrict__ wvt,
    const _Float16* __restrict__ wrt,   // [256][D] f16
    float* __restrict__ out) {
  extern __shared__ char smem[];
  _Float16* xh = (_Float16*)smem;

  const int tid  = threadIdx.x;
  const int wave = tid >> 5;
  const int lane = tid & 31;
  const int b    = blockIdx.x;

  // ---- stage x_b into LDS as f16 ----
  const float* xb = x + (size_t)b * (N_ * D_);
  for (int i = tid; i < (N_ * D_) / 4; i += 256) {
    const float4 v = ((const float4*)xb)[i];
    const int row = (i * 4) / D_;
    const int col = (i * 4) % D_;
    _Float16* p = xh + row * XSTR + col;
    p[0] = (_Float16)v.x; p[1] = (_Float16)v.y;
    p[2] = (_Float16)v.z; p[3] = (_Float16)v.w;
  }
  __syncthreads();

  const int h = wave;
  char* hb = smem + XH_BYTES + h * HEAD_BYTES;
  _Float16* Qh = (_Float16*)hb;
  _Float16* Kh = (_Float16*)(hb + QH_BYTES);
  _Float16* Vt = (_Float16*)(hb + QH_BYTES + KH_BYTES);
  float*    S  = (float*)(hb + QH_BYTES + KH_BYTES + VT_BYTES);

  const int nn = (lane & 15);
  const int mb = (lane >> 4) << 3;

  // ---- Q/K/V projections: (64x128)x(128x32) per head ----
  const _Float16* wmats[3] = { wqt + h * ATT_ * D_,
                               wkt + h * ATT_ * D_,
                               wvt + h * ATT_ * D_ };
  for (int mat = 0; mat < 3; ++mat) {
    v8f pacc[4][2] = {};
    const _Float16* w = wmats[mat];
#pragma unroll
    for (int kt = 0; kt < 4; ++kt) {
      v16h a0 = load_a_f16(xh, XSTR,  0, 32 * kt, lane);
      v16h a1 = load_a_f16(xh, XSTR, 16, 32 * kt, lane);
      v16h a2 = load_a_f16(xh, XSTR, 32, 32 * kt, lane);
      v16h a3 = load_a_f16(xh, XSTR, 48, 32 * kt, lane);
#pragma unroll
      for (int nt = 0; nt < 2; ++nt) {
        v16h bw = load_b_f16(w, D_, 16 * nt, 32 * kt, lane);
        pacc[0][nt] = wmma16(a0, bw, pacc[0][nt]);
        pacc[1][nt] = wmma16(a1, bw, pacc[1][nt]);
        pacc[2][nt] = wmma16(a2, bw, pacc[2][nt]);
        pacc[3][nt] = wmma16(a3, bw, pacc[3][nt]);
      }
    }
    // D layout: lane col N=nn, rows M = mb + r
#pragma unroll
    for (int mt = 0; mt < 4; ++mt)
#pragma unroll
      for (int nt = 0; nt < 2; ++nt) {
        const int col = 16 * nt + nn;
#pragma unroll
        for (int r = 0; r < 8; ++r) {
          const int row = 16 * mt + mb + r;
          const _Float16 val = (_Float16)pacc[mt][nt][r];
          if (mat == 0)      Qh[row * QSTR + col] = val;
          else if (mat == 1) Kh[row * QSTR + col] = val;
          else               Vt[col * VTSTR + row] = val;   // transposed for B use
        }
      }
  }
  __syncthreads();

  // ---- residual projection straight into output accumulators ----
  v8f acc[4][2] = {};
  {
    const _Float16* wr = wrt + (h * ATT_) * D_;
#pragma unroll
    for (int kt = 0; kt < 4; ++kt) {
      v16h a0 = load_a_f16(xh, XSTR,  0, 32 * kt, lane);
      v16h a1 = load_a_f16(xh, XSTR, 16, 32 * kt, lane);
      v16h a2 = load_a_f16(xh, XSTR, 32, 32 * kt, lane);
      v16h a3 = load_a_f16(xh, XSTR, 48, 32 * kt, lane);
#pragma unroll
      for (int nt = 0; nt < 2; ++nt) {
        v16h bw = load_b_f16(wr, D_, 16 * nt, 32 * kt, lane);
        acc[0][nt] = wmma16(a0, bw, acc[0][nt]);
        acc[1][nt] = wmma16(a1, bw, acc[1][nt]);
        acc[2][nt] = wmma16(a2, bw, acc[2][nt]);
        acc[3][nt] = wmma16(a3, bw, acc[3][nt]);
      }
    }
  }

  // ---- attention: one 16-row tile at a time (S buffer = 16x64 f32) ----
  for (int mt = 0; mt < 4; ++mt) {
    v16h aq = load_a_f16(Qh, QSTR, 16 * mt, 0, lane);   // K = 32, one tile
#pragma unroll
    for (int nt = 0; nt < 4; ++nt) {
      v16h bk = load_b_f16(Kh, QSTR, 16 * nt, 0, lane); // B = K^T via K rows
      v8f s = {};
      s = wmma16(aq, bk, s);
      const int col = 16 * nt + nn;
#pragma unroll
      for (int r = 0; r < 8; ++r) S[(mb + r) * SSTR + col] = s[r] * SCALE;
    }
    __syncthreads();

    if (lane < 16) {   // one lane per row, 64 columns
      float* row = S + lane * SSTR;
      float mx = row[0];
      for (int c = 1; c < 64; ++c) mx = fmaxf(mx, row[c]);
      float sum = 0.0f;
      for (int c = 0; c < 64; ++c) { float e = __expf(row[c] - mx); row[c] = e; sum += e; }
      const float inv = 1.0f / sum;
      for (int c = 0; c < 64; ++c) row[c] *= inv;
    }
    __syncthreads();

#pragma unroll
    for (int kt = 0; kt < 2; ++kt) {
      v16h aa = load_a_f32cvt(S, SSTR, 32 * kt, lane);      // attn rows (f32->f16)
#pragma unroll
      for (int nt = 0; nt < 2; ++nt) {
        v16h bv = load_b_f16(Vt, VTSTR, 16 * nt, 32 * kt, lane);
        acc[mt][nt] = wmma16(aa, bv, acc[mt][nt]);
      }
    }
    __syncthreads();   // S reused next iteration
  }

  // ---- relu(ctx + res) -> out[b][row][h*32 + col] ----
  float* ob = out + (size_t)b * (N_ * H_ * ATT_) + h * ATT_;
#pragma unroll
  for (int mt = 0; mt < 4; ++mt)
#pragma unroll
    for (int nt = 0; nt < 2; ++nt)
#pragma unroll
      for (int r = 0; r < 8; ++r) {
        const int row = 16 * mt + mb + r;
        ob[row * (H_ * ATT_) + 16 * nt + nn] = fmaxf(acc[mt][nt][r], 0.0f);
      }
}

extern "C" void kernel_launch(void* const* d_in, const int* in_sizes, int n_in,
                              void* d_out, int out_size, void* d_ws, size_t ws_size,
                              hipStream_t stream) {
  (void)in_sizes; (void)n_in; (void)out_size; (void)ws_size;
  const float* x  = (const float*)d_in[0];
  const float* WQ = (const float*)d_in[1];
  const float* WK = (const float*)d_in[2];
  const float* WV = (const float*)d_in[3];
  const float* WR = (const float*)d_in[4];
  float* out = (float*)d_out;

  _Float16* oq  = (_Float16*)d_ws;            // 32768 halves each
  _Float16* ok2 = oq  + 32768;
  _Float16* ov  = ok2 + 32768;
  _Float16* orr = ov  + 32768;

  prep_weights<<<128, 256, 0, stream>>>(WQ, WK, WV, WR, oq, ok2, ov, orr);

  hipFuncSetAttribute((const void*)attn_fused,
                      hipFuncAttributeMaxDynamicSharedMemorySize, SMEM_BYTES);
  attn_fused<<<4096, 256, SMEM_BYTES, stream>>>(x, oq, ok2, ov, orr, out);
}
